// MoELayer_22522808500374
// MI455X (gfx1250) — matrix operand order repaired and checked by reading, not compiled, not run on hardware
//
#include <hip/hip_runtime.h>

#define T_TOKENS 4096
#define D_EMBED  1024
#define F_HIDDEN 4096
#define N_EXPERTS 8
#define TILE_M   32
#define FCHUNK   512

typedef __bf16 v16bf __attribute__((ext_vector_type(16)));
typedef float  v8f   __attribute__((ext_vector_type(8)));
typedef unsigned int u32x4 __attribute__((ext_vector_type(4)));
typedef int          i32x8 __attribute__((ext_vector_type(8)));
typedef int          i32x4 __attribute__((ext_vector_type(4)));

// sA layout produced by TDM pad: 16B pad after every 1024B => row stride 2080B
// (1040 bf16). Row stride = 520 dwords == 8 mod 64 banks -> only 2-way conflicts.
#define SA_ROW_ELEMS 1040
__device__ __forceinline__ int sa_idx(int r, int c) {
    return r * SA_ROW_ELEMS + c + ((c >> 9) << 3);   // +8 elems past col 512
}
#define SA_BYTES (TILE_M * SA_ROW_ELEMS * 2)         // 66,560
#define H_STRIDE (FCHUNK + 8)
#define SH_BYTES (TILE_M * H_STRIDE * 2)             // 33,280
#define SMEM_BYTES (SA_BYTES + SH_BYTES + TILE_M * 4 + TILE_M * 4)

// Branch-free tanh-gelu: v_exp_f32 + v_rcp_f32, saturates via inf->rcp->0.
__device__ __forceinline__ float gelu_tanh(float x) {
    float u  = 0.7978845608028654f * (x + 0.044715f * x * x * x);
    float au = __builtin_fabsf(u);
    float t  = __builtin_amdgcn_exp2f(2.8853900817779268f * au);
    float th = 1.0f - 2.0f * __builtin_amdgcn_rcpf(t + 1.0f);
    th = __builtin_copysignf(th, u);
    return 0.5f * x * (1.0f + th);
}

// TDM gather: one descriptor moves 16 scattered token rows (1024 bf16 each)
// from xb into LDS at lds_addr, with 16B LDS pad per 1024B (D# section 8.3-8.7).
__device__ __forceinline__ void tdm_gather16(unsigned lds_addr, const __bf16* gbase,
                                             const int* tk) {
    unsigned long long ga = (unsigned long long)(uintptr_t)gbase;
    u32x4 g0;
    g0.x = 0x80000001u;                                 // gather_mode=1, idx16, count=1
    g0.y = lds_addr;                                    // D#.lds_addr
    g0.z = (unsigned)(ga & 0xFFFFFFFFull);              // global_addr[31:0]
    g0.w = (unsigned)((ga >> 32) & 0x01FFFFFFull) | (2u << 30);  // [56:32] | type=2
    i32x8 g1;
    g1[0] = (1 << 16) | (1 << 20) | (7 << 22) | (3 << 25); // 2B elts, pad, itv=256dw, amt=4dw
    g1[1] = (int)(1024u << 16);                         // tensor_dim0 = 1024 elts
    g1[2] = (int)(4096u << 16);                         // tensor_dim1 = 4096 rows
    g1[3] = (int)(1024u << 16);                         // tile_dim0 = 1024 elts
    g1[4] = 16;                                         // tile_dim1 = #indices
    g1[5] = 1024;                                       // tensor_dim0_stride
    g1[6] = 0; g1[7] = 0;
    i32x4 g2, g3;
#pragma unroll
    for (int p = 0; p < 4; ++p) {
        g2[p] = (tk[2 * p]     & 0xFFFF) | (tk[2 * p + 1] << 16);
        g3[p] = (tk[8 + 2 * p] & 0xFFFF) | (tk[9 + 2 * p] << 16);
    }
#if __clang_major__ >= 23
    i32x8 gz = {0, 0, 0, 0, 0, 0, 0, 0};
    __builtin_amdgcn_tensor_load_to_lds(g0, g1, g2, g3, gz, 0);
#else
    __builtin_amdgcn_tensor_load_to_lds(g0, g1, g2, g3, 0);
#endif
}

// ---------------------------------------------------------------------------
// Kernel 0: zero output + routing counters, convert x fp32 -> bf16
// ---------------------------------------------------------------------------
__global__ __launch_bounds__(256) void init_kernel(
    const float* __restrict__ x, __bf16* __restrict__ xb,
    float* __restrict__ out, int* __restrict__ cnt, int n) {
    int i = blockIdx.x * blockDim.x + threadIdx.x;
    if (i < N_EXPERTS) cnt[i] = 0;
    if (i < n) {
        out[i] = 0.0f;
        xb[i]  = (__bf16)x[i];
    }
}

// ---------------------------------------------------------------------------
// Kernel 1: gating -> combine weights + compacted per-expert token lists
// ---------------------------------------------------------------------------
__global__ __launch_bounds__(256) void gate_kernel(
    const float* __restrict__ x, const float* __restrict__ Wg,
    const float* __restrict__ bg, float* __restrict__ cw,
    int* __restrict__ cnt, int* __restrict__ idxl) {
    int t = blockIdx.x * blockDim.x + threadIdx.x;
    if (t >= T_TOKENS) return;

    float logit[N_EXPERTS];
#pragma unroll
    for (int e = 0; e < N_EXPERTS; ++e) logit[e] = bg[e];
    const float* xr = x + (size_t)t * D_EMBED;
#pragma unroll 4
    for (int k = 0; k < D_EMBED; ++k) {
        float xv = xr[k];
        const float* wr = Wg + k * N_EXPERTS;
#pragma unroll
        for (int e = 0; e < N_EXPERTS; ++e) logit[e] = fmaf(xv, wr[e], logit[e]);
    }
    float m = logit[0];
#pragma unroll
    for (int e = 1; e < N_EXPERTS; ++e) m = fmaxf(m, logit[e]);
    float p[N_EXPERTS]; float sum = 0.0f;
#pragma unroll
    for (int e = 0; e < N_EXPERTS; ++e) { p[e] = __expf(logit[e] - m); sum += p[e]; }

    float v0 = -1.0f, v1 = -1.0f; int i0 = 0, i1 = 0;
#pragma unroll
    for (int e = 0; e < N_EXPERTS; ++e) {
        float pe = p[e];
        if (pe > v0)      { v1 = v0; i1 = i0; v0 = pe; i0 = e; }
        else if (pe > v1) { v1 = pe; i1 = e; }
    }
    float p0 = v0 / sum, p1 = v1 / sum;
    float inv = 1.0f / (p0 + p1 + 1e-8f);
#pragma unroll
    for (int e = 0; e < N_EXPERTS; ++e)
        cw[t * N_EXPERTS + e] = (e == i0) ? p0 * inv : ((e == i1) ? p1 * inv : 0.0f);

    int s0 = atomicAdd(&cnt[i0], 1);
    idxl[i0 * T_TOKENS + s0] = t;
    int s1 = atomicAdd(&cnt[i1], 1);
    idxl[i1 * T_TOKENS + s1] = t;
}

// ---------------------------------------------------------------------------
// Kernel 2: routed expert FFN. grid=(max tiles, experts); 256 thr = 8 waves.
// TDM-gathers 32 routed tokens, runs fused W1/gelu/W2, scatter-adds output.
// ---------------------------------------------------------------------------
__global__ __launch_bounds__(256) void moe_ffn_kernel(
    const __bf16* __restrict__ xb, const float* __restrict__ cw,
    const int* __restrict__ cnt, const int* __restrict__ idxl,
    const float* __restrict__ W1, const float* __restrict__ b1,
    const float* __restrict__ W2, const float* __restrict__ b2,
    float* __restrict__ out) {
    extern __shared__ char smem[];
    __bf16* sA   = (__bf16*)smem;                       // TDM-padded x gather
    __bf16* sH   = (__bf16*)(smem + SA_BYTES);          // hidden chunk
    float*  sW   = (float*)(smem + SA_BYTES + SH_BYTES);
    int*    sTok = (int*)(smem + SA_BYTES + SH_BYTES + TILE_M * 4);

    const int tile  = blockIdx.x;
    const int e     = blockIdx.y;
    const int tid   = threadIdx.x;
    const int wave  = tid >> 5;
    const int lane  = tid & 31;
    const int lhalf = lane >> 4;
    const int l16   = lane & 15;

    const int count = cnt[e];
    if (tile * TILE_M >= count) return;               // routed-only: skip empty tiles
    const int rem = count - tile * TILE_M;            // 1..TILE_M valid rows
    const int* lst = idxl + e * T_TOKENS + tile * TILE_M;

    if (tid < TILE_M) {
        int token = lst[tid < rem ? tid : 0];         // clamp pads to a valid token
        sTok[tid] = token;
        sW[tid]   = (tid < rem) ? cw[token * N_EXPERTS + e] : 0.0f;
    }

    // ---- TDM gather of 32 token rows (2 descriptors), wave 0 only ----
    if (wave == 0) {
        int tk[TILE_M];
#pragma unroll
        for (int k = 0; k < TILE_M; ++k) tk[k] = lst[k < rem ? k : 0];
        const unsigned lds0 = (unsigned)(uintptr_t)(void*)sA;
        tdm_gather16(lds0, xb, tk);
        tdm_gather16(lds0 + 16u * SA_ROW_ELEMS * 2u, xb, tk + 16);
        __builtin_amdgcn_s_wait_tensorcnt(0);
    }
    __syncthreads();

    const float* W1e = W1 + (size_t)e * D_EMBED * F_HIDDEN;
    const float* W2e = W2 + (size_t)e * F_HIDDEN * D_EMBED;
    const float* b1e = b1 + e * F_HIDDEN;
    const float* b2e = b2 + e * D_EMBED;

    v8f acc[16];
    const v8f vzero = {0.f, 0.f, 0.f, 0.f, 0.f, 0.f, 0.f, 0.f};
#pragma unroll
    for (int j = 0; j < 16; ++j) acc[j] = vzero;

#pragma unroll 1
    for (int fc = 0; fc < F_HIDDEN / FCHUNK; ++fc) {
        __builtin_prefetch(W2e + (size_t)(fc * FCHUNK + tid * 2) * D_EMBED, 0, 1);

        // ---- Stage 1: 4 n-subtiles/wave x 2 m-tiles, shared B fragment ----
#pragma unroll 1
        for (int s = 0; s < 4; ++s) {
            const int nloc = (wave * 4 + s) * 16;
            const int nglb = fc * FCHUNK + nloc;
            v8f c0 = vzero, c1 = vzero;
            const float* bcol = W1e + (nglb + l16);
#pragma unroll 1
            for (int k0 = 0; k0 < D_EMBED; k0 += 32) {
                v16bf a0, a1, b;
                const __bf16* ar0 = &sA[sa_idx(l16,      k0 + lhalf * 8)];
                const __bf16* ar1 = &sA[sa_idx(16 + l16, k0 + lhalf * 8)];
#pragma unroll
                for (int i = 0; i < 8; ++i) {
                    a0[i] = ar0[i]; a0[i + 8] = ar0[16 + i];
                    a1[i] = ar1[i]; a1[i + 8] = ar1[16 + i];
                }
#pragma unroll
                for (int q = 0; q < 16; ++q)
                    b[q] = (__bf16)bcol[(size_t)(k0 + lhalf * 8 + (q & 7) + ((q >> 3) << 4)) * F_HIDDEN];
                c0 = __builtin_amdgcn_wmma_f32_16x16x32_bf16(
                         false, a0, false, b, (short)0, c0, false, false);
                c1 = __builtin_amdgcn_wmma_f32_16x16x32_bf16(
                         false, a1, false, b, (short)0, c1, false, false);
            }
            const float bias = b1e[nglb + l16];
#pragma unroll
            for (int r = 0; r < 8; ++r) {
                const int rowm = lhalf * 8 + r;
                sH[rowm * H_STRIDE + nloc + l16]        = (__bf16)gelu_tanh(c0[r] + bias);
                sH[(16 + rowm) * H_STRIDE + nloc + l16] = (__bf16)gelu_tanh(c1[r] + bias);
            }
        }
        __syncthreads();

        // ---- Stage 2: fold chunk into per-wave 32x128 accumulators ----
#pragma unroll 1
        for (int k0 = 0; k0 < FCHUNK; k0 += 32) {
            v16bf a0, a1;
            const __bf16* hr0 = &sH[l16 * H_STRIDE + k0 + lhalf * 8];
            const __bf16* hr1 = &sH[(16 + l16) * H_STRIDE + k0 + lhalf * 8];
#pragma unroll
            for (int i = 0; i < 8; ++i) {
                a0[i] = hr0[i]; a0[i + 8] = hr0[16 + i];
                a1[i] = hr1[i]; a1[i + 8] = hr1[16 + i];
            }
            const int kglb = fc * FCHUNK + k0 + lhalf * 8;
#pragma unroll
            for (int j = 0; j < 8; ++j) {
                const int n = wave * 128 + j * 16 + l16;
                v16bf b;
#pragma unroll
                for (int q = 0; q < 16; ++q)
                    b[q] = (__bf16)W2e[(size_t)(kglb + (q & 7) + ((q >> 3) << 4)) * D_EMBED + n];
                acc[2 * j]     = __builtin_amdgcn_wmma_f32_16x16x32_bf16(
                                     false, a0, false, b, (short)0, acc[2 * j], false, false);
                acc[2 * j + 1] = __builtin_amdgcn_wmma_f32_16x16x32_bf16(
                                     false, a1, false, b, (short)0, acc[2 * j + 1], false, false);
            }
        }
        __syncthreads();
    }

    // ---- Epilogue: bias, combine weight, scatter-add per routed token ----
#pragma unroll
    for (int j = 0; j < 8; ++j) {
        const int n = wave * 128 + j * 16 + l16;
        const float bias = b2e[n];
#pragma unroll
        for (int mt = 0; mt < 2; ++mt) {
#pragma unroll
            for (int r = 0; r < 8; ++r) {
                const int rowm = mt * 16 + lhalf * 8 + r;
                const float wgt = sW[rowm];
                if (wgt != 0.0f) {
                    unsafeAtomicAdd(&out[(size_t)sTok[rowm] * D_EMBED + n],
                                    (acc[2 * j + mt][r] + bias) * wgt);
                }
            }
        }
    }
}

// ---------------------------------------------------------------------------
extern "C" void kernel_launch(void* const* d_in, const int* in_sizes, int n_in,
                              void* d_out, int out_size, void* d_ws, size_t ws_size,
                              hipStream_t stream) {
    (void)in_sizes; (void)n_in; (void)out_size; (void)ws_size;
    const float* x  = (const float*)d_in[0];
    const float* Wg = (const float*)d_in[1];
    const float* bg = (const float*)d_in[2];
    const float* W1 = (const float*)d_in[3];
    const float* b1 = (const float*)d_in[4];
    const float* W2 = (const float*)d_in[5];
    const float* b2 = (const float*)d_in[6];
    float* out = (float*)d_out;

    char* ws = (char*)d_ws;
    __bf16* xb  = (__bf16*)ws;                                   // 8 MB
    float*  cw  = (float*)(ws + (size_t)T_TOKENS * D_EMBED * 2); // 128 KB
    int*    cnt = (int*)(ws + (size_t)T_TOKENS * D_EMBED * 2 + T_TOKENS * N_EXPERTS * 4);
    int*    idxl = cnt + 32;                                     // 8*4096 ints

    const int n = T_TOKENS * D_EMBED;
    init_kernel<<<(n + 255) / 256, 256, 0, stream>>>(x, xb, out, cnt, n);
    gate_kernel<<<(T_TOKENS + 255) / 256, 256, 0, stream>>>(x, Wg, bg, cw, cnt, idxl);

    hipFuncSetAttribute((const void*)moe_ffn_kernel,
                        hipFuncAttributeMaxDynamicSharedMemorySize, SMEM_BYTES);
    dim3 grid(T_TOKENS / TILE_M, N_EXPERTS);   // x fastest -> expert-major L2 reuse
    moe_ffn_kernel<<<grid, 256, SMEM_BYTES, stream>>>(xb, cw, cnt, idxl,
                                                      W1, b1, W2, b2, out);
}
